// Mamba_55911884259813
// MI455X (gfx1250) — matrix operand order repaired and checked by reading, not compiled, not run on hardware
//
#include <hip/hip_runtime.h>

// ---------------------------------------------------------------------------
// Mamba block forward for MI455X (gfx1250, wave32, WMMA).
//   Large GEMMs: bf16 V_WMMA_F32_16X16X32_BF16, 32x64 C-tile per wave
//                (8 WMMAs / K-step, ~21 FLOP/byte from L2), 128x128 per block.
//   Small GEMM (N=96): 16x16-per-wave variant.
//   Scan: chunked parallel scan (64-step chunks) exploiting
//         prod_t exp(dt_t*A) == exp(A * sum_t dt_t); serial chain 2048 -> 160.
// ---------------------------------------------------------------------------

#define B_SZ    2
#define SEQ_L   2048
#define DMODEL  1024
#define DINNER  2048
#define DSTATE  16
#define DCONV   4
#define DTRANK  64
#define NXPROJ  (DTRANK + 2 * DSTATE)   // 96
#define NROWS   (B_SZ * SEQ_L)          // 4096
#define TCHUNK  64
#define NCHUNK  (SEQ_L / TCHUNK)        // 32

typedef __attribute__((ext_vector_type(16))) __bf16 v16bf;
typedef __attribute__((ext_vector_type(8)))  __bf16 v8bf;
typedef __attribute__((ext_vector_type(8)))  float  v8f;

// ---------------- f32 -> bf16 conversions ----------------------------------
__global__ void cvt_f32_bf16(const float* __restrict__ src,
                             __bf16* __restrict__ dst, long long n) {
  long long i = (long long)blockIdx.x * blockDim.x + threadIdx.x;
  if (i < n) dst[i] = (__bf16)src[i];
}

__global__ void cvt_strided_f32_bf16(const float* __restrict__ src,
                                     __bf16* __restrict__ dst,
                                     int rows, int cols, int src_stride) {
  int i = blockIdx.x * blockDim.x + threadIdx.x;
  if (i >= rows * cols) return;
  int r = i / cols, c = i - r * cols;
  dst[i] = (__bf16)src[(long long)r * src_stride + c];
}

// ---------------- WMMA fragment helpers ------------------------------------
// A 16x32 frag: lane l<16 -> row l,  K 0..15 contiguous; l>=16 -> row l-16, K 16..31.
// B 32x16 frag: lane l<16 -> col l,  K 0..15 contiguous; l>=16 -> K 16..31.
__device__ __forceinline__ v16bf load_a_frag(const __bf16* p) {
  v8bf lo = *(const v8bf*)(p);
  v8bf hi = *(const v8bf*)(p + 8);
  return __builtin_shufflevector(lo, hi, 0,1,2,3,4,5,6,7,8,9,10,11,12,13,14,15);
}

// ---------------- fat bf16 WMMA GEMM: C[M,N] = A[M,K] @ B[N,K]^T -----------
// 256 threads = 8 waves; wave w: wm=w&3 (32-row slab), wn=w>>2 (64-col slab);
// block tile 128x128. Requires M%128==0, N%128==0, K%32==0.
__global__ __launch_bounds__(256)
void gemm_bf16_wmma_fat(const __bf16* __restrict__ A, const __bf16* __restrict__ Bw,
                        float* __restrict__ C, int M, int N, int K) {
  const int lane = threadIdx.x & 31;
  const int wave = threadIdx.x >> 5;
  const int m0 = blockIdx.y * 128 + (wave & 3) * 32;
  const int n0 = blockIdx.x * 128 + (wave >> 2) * 64;

  const int ak = (lane >> 4) * 16;    // A K sub-offset (0 / 16)
  const int bk = (lane >> 4) * 16;    // B K sub-offset (0 / 16)
  const __bf16* ap0 = A  + (size_t)(m0      + (lane & 15)) * K + ak;
  const __bf16* ap1 = A  + (size_t)(m0 + 16 + (lane & 15)) * K + ak;
  const __bf16* bp0 = Bw + (size_t)(n0      + (lane & 15)) * K + bk;
  const __bf16* bp1 = Bw + (size_t)(n0 + 16 + (lane & 15)) * K + bk;
  const __bf16* bp2 = Bw + (size_t)(n0 + 32 + (lane & 15)) * K + bk;
  const __bf16* bp3 = Bw + (size_t)(n0 + 48 + (lane & 15)) * K + bk;

  v8f acc[2][4];
#pragma unroll
  for (int i = 0; i < 2; ++i)
#pragma unroll
    for (int j = 0; j < 4; ++j) acc[i][j] = (v8f){};

  for (int k = 0; k < K; k += 32) {
    v16bf a0 = load_a_frag(ap0 + k);
    v16bf a1 = load_a_frag(ap1 + k);
    v16bf b0 = *(const v16bf*)(bp0 + k);
    v16bf b1 = *(const v16bf*)(bp1 + k);
    v16bf b2 = *(const v16bf*)(bp2 + k);
    v16bf b3 = *(const v16bf*)(bp3 + k);
    acc[0][0] = __builtin_amdgcn_wmma_f32_16x16x32_bf16(false, a0, false, b0, (short)0, acc[0][0], false, false);
    acc[0][1] = __builtin_amdgcn_wmma_f32_16x16x32_bf16(false, a0, false, b1, (short)0, acc[0][1], false, false);
    acc[0][2] = __builtin_amdgcn_wmma_f32_16x16x32_bf16(false, a0, false, b2, (short)0, acc[0][2], false, false);
    acc[0][3] = __builtin_amdgcn_wmma_f32_16x16x32_bf16(false, a0, false, b3, (short)0, acc[0][3], false, false);
    acc[1][0] = __builtin_amdgcn_wmma_f32_16x16x32_bf16(false, a1, false, b0, (short)0, acc[1][0], false, false);
    acc[1][1] = __builtin_amdgcn_wmma_f32_16x16x32_bf16(false, a1, false, b1, (short)0, acc[1][1], false, false);
    acc[1][2] = __builtin_amdgcn_wmma_f32_16x16x32_bf16(false, a1, false, b2, (short)0, acc[1][2], false, false);
    acc[1][3] = __builtin_amdgcn_wmma_f32_16x16x32_bf16(false, a1, false, b3, (short)0, acc[1][3], false, false);
  }

  // C 16x16: VGPR r -> (M = base + (lane>=16 ? 8:0) + r, N = base + lane&15)
  const int cn = (lane & 15);
  const int cm = (lane >> 4) ? 8 : 0;
#pragma unroll
  for (int i = 0; i < 2; ++i)
#pragma unroll
    for (int j = 0; j < 4; ++j) {
      float* cp = C + (size_t)(m0 + 16 * i + cm) * N + (n0 + 16 * j + cn);
#pragma unroll
      for (int r = 0; r < 8; ++r) cp[(size_t)r * N] = acc[i][j][r];
    }
}

// ---------------- simple GEMM for small N (x_proj, N=96) -------------------
__global__ __launch_bounds__(256)
void gemm_bf16_wmma(const __bf16* __restrict__ A, const __bf16* __restrict__ Bw,
                    float* __restrict__ C, int M, int N, int K) {
  const int lane = threadIdx.x & 31;
  const int wave = threadIdx.x >> 5;
  const int n0 = blockIdx.x * 16;
  const int m0 = (blockIdx.y * 8 + wave) * 16;
  if (m0 >= M || n0 >= N) return;

  const __bf16* ap = A  + (size_t)(m0 + (lane & 15)) * K + (lane >> 4) * 16;
  const __bf16* bp = Bw + (size_t)(n0 + (lane & 15)) * K + (lane >> 4) * 16;

  v8f c = {};
  for (int k = 0; k < K; k += 32) {
    v16bf av = load_a_frag(ap + k);
    v16bf bv = *(const v16bf*)(bp + k);
    c = __builtin_amdgcn_wmma_f32_16x16x32_bf16(false, av, false, bv, (short)0, c, false, false);
  }
  const int cn = n0 + (lane & 15);
  const int cm = m0 + ((lane >> 4) ? 8 : 0);
#pragma unroll
  for (int r = 0; r < 8; ++r) C[(size_t)(cm + r) * N + cn] = c[r];
}

// ---------------- causal depthwise conv (d_conv=4) + SiLU ------------------
__global__ void conv_silu(const float* __restrict__ xz,
                          const float* __restrict__ cw,
                          const float* __restrict__ cb,
                          float* __restrict__ x_f32,
                          __bf16* __restrict__ x_bf) {
  long long i = (long long)blockIdx.x * blockDim.x + threadIdx.x;
  if (i >= (long long)NROWS * DINNER) return;
  int d = (int)(i % DINNER);
  long long row = i / DINNER;
  int l = (int)(row % SEQ_L);
  long long base = row * (2 * DINNER);
  float acc = cb[d];
#pragma unroll
  for (int j = 0; j < DCONV; ++j) {
    int lt = l - (DCONV - 1) + j;
    if (lt >= 0)
      acc += xz[base + (long long)(lt - l) * (2 * DINNER) + d] * cw[d * DCONV + j];
  }
  float s = acc / (1.0f + __expf(-acc));
  x_f32[i] = s;
  x_bf[i]  = (__bf16)s;
}

__device__ __forceinline__ float softplusf(float v) {
  return (v > 20.0f) ? v : log1pf(__expf(v));
}

// ---------------- chunked scan, pass 1: per-chunk partials -----------------
// thread = (b, d, chunk); computes chunk-local state S (zero init) and
// sum(dt) over the chunk (chunk decay = exp(A[n] * sumdt)).
__global__ __launch_bounds__(64)
void scan_part1(const float* __restrict__ dt_full, const float* __restrict__ x_f32,
                const float* __restrict__ x_dbl,   const float* __restrict__ dt_b,
                const float* __restrict__ A_log,
                float* __restrict__ Sbuf, float* __restrict__ sumdt_buf) {
  const int dgrp = blockIdx.x & 31;                 // DINNER/64 groups
  const int c    = (blockIdx.x >> 5) & (NCHUNK - 1);
  const int b    = blockIdx.x >> 10;
  const int d    = dgrp * 64 + threadIdx.x;
  __shared__ float sB[DSTATE];

  float A[DSTATE], S[DSTATE];
#pragma unroll
  for (int n = 0; n < DSTATE; ++n) {
    A[n] = -__expf(A_log[d * DSTATE + n]);
    S[n] = 0.0f;
  }
  const float bias = dt_b[d];
  float sumdt = 0.0f;

  for (int tt = 0; tt < TCHUNK; ++tt) {
    const long long row = (long long)b * SEQ_L + c * TCHUNK + tt;
    if (threadIdx.x < DSTATE)
      sB[threadIdx.x] = x_dbl[row * NXPROJ + DTRANK + threadIdx.x];
    __syncthreads();
    float dtv = softplusf(dt_full[row * DINNER + d] + bias);
    float xv  = x_f32[row * DINNER + d];
    sumdt += dtv;
#pragma unroll
    for (int n = 0; n < DSTATE; ++n)
      S[n] = S[n] * __expf(dtv * A[n]) + (xv * dtv) * sB[n];
    __syncthreads();
  }

  const size_t off = ((size_t)b * DINNER + d) * NCHUNK + c;
#pragma unroll
  for (int n = 0; n < DSTATE; ++n) Sbuf[off * DSTATE + n] = S[n];
  sumdt_buf[off] = sumdt;
}

// ---------------- chunked scan, carry combine ------------------------------
// thread = (b,d); 32-step sequential combine over chunks:
//   carry[c] = exp(A*sumdt[c-1]) * carry[c-1] + S[c-1], carry[0] = 0.
__global__ __launch_bounds__(256)
void scan_combine(const float* __restrict__ Sbuf, const float* __restrict__ sumdt_buf,
                  const float* __restrict__ A_log, float* __restrict__ carry) {
  const int i = blockIdx.x * 256 + threadIdx.x;     // b*DINNER + d
  const int d = i & (DINNER - 1);
  float A[DSTATE], st[DSTATE];
#pragma unroll
  for (int n = 0; n < DSTATE; ++n) {
    A[n]  = -__expf(A_log[d * DSTATE + n]);
    st[n] = 0.0f;
  }
  for (int c = 0; c < NCHUNK; ++c) {
    const size_t off = (size_t)i * NCHUNK + c;
#pragma unroll
    for (int n = 0; n < DSTATE; ++n) carry[off * DSTATE + n] = st[n];
    const float sd = sumdt_buf[off];
#pragma unroll
    for (int n = 0; n < DSTATE; ++n)
      st[n] = st[n] * __expf(A[n] * sd) + Sbuf[off * DSTATE + n];
  }
}

// ---------------- chunked scan, pass 2: emit outputs -----------------------
__global__ __launch_bounds__(64)
void scan_part2(const float* __restrict__ dt_full, const float* __restrict__ x_f32,
                const float* __restrict__ x_dbl,   const float* __restrict__ xz,
                const float* __restrict__ dt_b,    const float* __restrict__ A_log,
                const float* __restrict__ Dco,     const float* __restrict__ carry,
                __bf16* __restrict__ y_bf) {
  const int dgrp = blockIdx.x & 31;
  const int c    = (blockIdx.x >> 5) & (NCHUNK - 1);
  const int b    = blockIdx.x >> 10;
  const int d    = dgrp * 64 + threadIdx.x;
  __shared__ float sBC[2 * DSTATE];

  float A[DSTATE], st[DSTATE];
  const size_t off = ((size_t)b * DINNER + d) * NCHUNK + c;
#pragma unroll
  for (int n = 0; n < DSTATE; ++n) {
    A[n]  = -__expf(A_log[d * DSTATE + n]);
    st[n] = carry[off * DSTATE + n];
  }
  const float bias = dt_b[d];
  const float Dd   = Dco[d];

  for (int tt = 0; tt < TCHUNK; ++tt) {
    const long long row = (long long)b * SEQ_L + c * TCHUNK + tt;
    if (threadIdx.x < 2 * DSTATE)
      sBC[threadIdx.x] = x_dbl[row * NXPROJ + DTRANK + threadIdx.x];
    __syncthreads();
    float dtv = softplusf(dt_full[row * DINNER + d] + bias);
    float xv  = x_f32[row * DINNER + d];
    float zv  = xz[row * (2 * DINNER) + DINNER + d];
    float acc = 0.0f;
#pragma unroll
    for (int n = 0; n < DSTATE; ++n) {
      st[n] = st[n] * __expf(dtv * A[n]) + (xv * dtv) * sBC[n];
      acc  += st[n] * sBC[DSTATE + n];
    }
    acc += Dd * xv;
    float out = acc * (zv / (1.0f + __expf(-zv)));
    y_bf[row * DINNER + d] = (__bf16)out;
    __syncthreads();
  }
}

// ---------------------------------------------------------------------------
extern "C" void kernel_launch(void* const* d_in, const int* in_sizes, int n_in,
                              void* d_out, int out_size, void* d_ws, size_t ws_size,
                              hipStream_t stream) {
  const float* hs    = (const float*)d_in[0];
  const float* w_in  = (const float*)d_in[1];
  const float* cw    = (const float*)d_in[2];
  const float* cb    = (const float*)d_in[3];
  const float* w_xp  = (const float*)d_in[4];
  const float* w_dt  = (const float*)d_in[5];
  const float* dt_b  = (const float*)d_in[6];
  const float* A_log = (const float*)d_in[7];
  const float* Dco   = (const float*)d_in[8];
  const float* w_out = (const float*)d_in[9];
  float* out = (float*)d_out;

  char* w = (char*)d_ws;
  auto alloc = [&](size_t bytes) {
    void* p = (void*)w; w += (bytes + 255) & ~(size_t)255; return p;
  };
  __bf16* hs_bf    = (__bf16*)alloc((size_t)NROWS * DMODEL * 2);
  __bf16* win_bf   = (__bf16*)alloc((size_t)2 * DINNER * DMODEL * 2);
  __bf16* wxp_bf   = (__bf16*)alloc((size_t)NXPROJ * DINNER * 2);
  __bf16* wdt_bf   = (__bf16*)alloc((size_t)DINNER * DTRANK * 2);
  __bf16* wout_bf  = (__bf16*)alloc((size_t)DMODEL * DINNER * 2);
  float*  xz       = (float*)alloc((size_t)NROWS * 2 * DINNER * 4);
  float*  x_f32    = (float*)alloc((size_t)NROWS * DINNER * 4);
  __bf16* x_bf     = (__bf16*)alloc((size_t)NROWS * DINNER * 2);
  float*  x_dbl    = (float*)alloc((size_t)NROWS * NXPROJ * 4);
  __bf16* dtin_bf  = (__bf16*)alloc((size_t)NROWS * DTRANK * 2);
  float*  dt_full  = (float*)alloc((size_t)NROWS * DINNER * 4);
  __bf16* y_bf     = (__bf16*)alloc((size_t)NROWS * DINNER * 2);
  float*  Sbuf     = (float*)alloc((size_t)B_SZ * DINNER * NCHUNK * DSTATE * 4);
  float*  carry    = (float*)alloc((size_t)B_SZ * DINNER * NCHUNK * DSTATE * 4);
  float*  sumdt    = (float*)alloc((size_t)B_SZ * DINNER * NCHUNK * 4);

  auto cdiv = [](long long a, long long b) { return (int)((a + b - 1) / b); };

  // bf16 conversions
  cvt_f32_bf16<<<cdiv((long long)NROWS * DMODEL, 256), 256, 0, stream>>>(
      hs, hs_bf, (long long)NROWS * DMODEL);
  cvt_f32_bf16<<<cdiv((long long)2 * DINNER * DMODEL, 256), 256, 0, stream>>>(
      w_in, win_bf, (long long)2 * DINNER * DMODEL);
  cvt_f32_bf16<<<cdiv((long long)NXPROJ * DINNER, 256), 256, 0, stream>>>(
      w_xp, wxp_bf, (long long)NXPROJ * DINNER);
  cvt_f32_bf16<<<cdiv((long long)DINNER * DTRANK, 256), 256, 0, stream>>>(
      w_dt, wdt_bf, (long long)DINNER * DTRANK);
  cvt_f32_bf16<<<cdiv((long long)DMODEL * DINNER, 256), 256, 0, stream>>>(
      w_out, wout_bf, (long long)DMODEL * DINNER);

  // xz = hs @ in_proj_w.T            (4096 x 4096, K=1024)
  gemm_bf16_wmma_fat<<<dim3(2 * DINNER / 128, NROWS / 128), 256, 0, stream>>>(
      hs_bf, win_bf, xz, NROWS, 2 * DINNER, DMODEL);

  // causal depthwise conv + SiLU
  conv_silu<<<cdiv((long long)NROWS * DINNER, 256), 256, 0, stream>>>(
      xz, cw, cb, x_f32, x_bf);

  // x_dbl = x_bl @ x_proj_w.T        (4096 x 96, K=2048)
  gemm_bf16_wmma<<<dim3(NXPROJ / 16, NROWS / 128), 256, 0, stream>>>(
      x_bf, wxp_bf, x_dbl, NROWS, NXPROJ, DINNER);

  // dt slice -> bf16
  cvt_strided_f32_bf16<<<cdiv((long long)NROWS * DTRANK, 256), 256, 0, stream>>>(
      x_dbl, dtin_bf, NROWS, DTRANK, NXPROJ);

  // dt_full = dt_in @ dt_proj_w.T    (4096 x 2048, K=64)
  gemm_bf16_wmma_fat<<<dim3(DINNER / 128, NROWS / 128), 256, 0, stream>>>(
      dtin_bf, wdt_bf, dt_full, NROWS, DINNER, DTRANK);

  // chunked parallel scan
  const int scan_blocks = B_SZ * NCHUNK * (DINNER / 64);   // 2048
  scan_part1<<<dim3(scan_blocks), 64, 0, stream>>>(
      dt_full, x_f32, x_dbl, dt_b, A_log, Sbuf, sumdt);
  scan_combine<<<dim3(B_SZ * DINNER / 256), 256, 0, stream>>>(
      Sbuf, sumdt, A_log, carry);
  scan_part2<<<dim3(scan_blocks), 64, 0, stream>>>(
      dt_full, x_f32, x_dbl, xz, dt_b, A_log, Dco, carry, y_bf);

  // out = y @ out_proj_w.T           (4096 x 1024, K=2048) -> f32 d_out
  gemm_bf16_wmma_fat<<<dim3(DMODEL / 128, NROWS / 128), 256, 0, stream>>>(
      y_bf, wout_bf, out, NROWS, DMODEL, DINNER);
}
